// HeteroGAT_395136991467
// MI455X (gfx1250) — matrix operand order repaired and checked by reading, not compiled, not run on hardware
//
#include <hip/hip_runtime.h>

typedef __attribute__((ext_vector_type(16))) __bf16 v16bf;
typedef __attribute__((ext_vector_type(8)))  float  v8f;

union Frag { v16bf v; float4 f[2]; };

#define BM 128
#define BN 128
#define BKK 32
#define APAD 8
#define BPAD 8
#define ALD (BKK + APAD)
#define BLD (BKK + BPAD)

// ---- CDNA5 async global->LDS path (guarded so either toolchain compiles) ----
#if defined(__has_builtin)
#if __has_builtin(__builtin_amdgcn_global_load_async_to_lds_b128)
#define HAVE_ASYNC_LDS 1
#endif
#endif
#ifndef HAVE_ASYNC_LDS
#define HAVE_ASYNC_LDS 0
#endif

#if HAVE_ASYNC_LDS
typedef int vi4 __attribute__((vector_size(16)));
#if __has_builtin(__builtin_amdgcn_s_wait_asynccnt)
#define ASYNC_WAIT() __builtin_amdgcn_s_wait_asynccnt(0)
#else
#define ASYNC_WAIT() asm volatile("s_wait_asynccnt 0" ::: "memory")
#endif
__device__ inline void async_cp16(const __bf16* g, __bf16* l) {
    __builtin_amdgcn_global_load_async_to_lds_b128(
        (__attribute__((address_space(1))) vi4*)(g),
        (__attribute__((address_space(3))) vi4*)(l), 0, 0);
}
#else
#define ASYNC_WAIT() do {} while (0)
#endif

static inline int cdiv(int a, int b) { return (a + b - 1) / b; }

__device__ inline __bf16 bf16_zero() {
    unsigned short z = 0;
    return __builtin_bit_cast(__bf16, z);
}

__device__ inline __bf16 to_bf16(float f) {
    unsigned int u = __float_as_uint(f);
    unsigned int r = (u + 0x7FFFu + ((u >> 16) & 1u)) >> 16;
    unsigned short s = (unsigned short)r;
    return __builtin_bit_cast(__bf16, s);
}

// guarded 8x bf16 (16B) load, zero-filled past K
__device__ inline float4 ldg8_guard(const __bf16* __restrict__ p, int k, int K) {
    if (k + 8 <= K) return *(const float4*)p;
    __align__(16) __bf16 tmp[8];
#pragma unroll
    for (int i = 0; i < 8; ++i) tmp[i] = (k + i < K) ? p[i] : bf16_zero();
    return *(const float4*)tmp;
}

// stage one 16B chunk into LDS: async DMA when fully in-bounds, else VGPR path
__device__ inline void stage16(const __bf16* g, bool inb, int k, int K, __bf16* l) {
#if HAVE_ASYNC_LDS
    if (inb && (k + 8 <= K)) { async_cp16(g, l); return; }
#endif
    float4 v = make_float4(0.f, 0.f, 0.f, 0.f);
    if (inb) v = ldg8_guard(g, k, K);
    *(float4*)l = v;
}

// stage one 16B chunk known aligned/full (only row-validity guard)
__device__ inline void stage16_full(const __bf16* g, bool inb, __bf16* l) {
#if HAVE_ASYNC_LDS
    if (inb) { async_cp16(g, l); return; }
    *(float4*)l = make_float4(0.f, 0.f, 0.f, 0.f);
#else
    float4 v = make_float4(0.f, 0.f, 0.f, 0.f);
    if (inb) v = *(const float4*)g;
    *(float4*)l = v;
#endif
}

// ---------------------------------------------------------------------------
// Generic bf16 WMMA GEMM: C[M,N] = A[M,K] * Bt[N,K]^T  (A bf16 row-major,
// Bt = B transposed, bf16 row-major), f32 accumulate. Double-buffered LDS,
// async global->LDS copies overlap with v_wmma compute.
// EPI: 0 = none, 1 = +bias, 2 = +bias then leaky_relu(0.01)
// ---------------------------------------------------------------------------
template <int EPI>
__global__ __launch_bounds__(256) void gemm_bf16_kernel(
    const __bf16* __restrict__ A, const __bf16* __restrict__ Bt,
    float* __restrict__ C, int M, int N, int K, const float* __restrict__ bias) {
    __shared__ __bf16 As[2][BM * ALD];
    __shared__ __bf16 Bs[2][BN * BLD];

    const int tid  = threadIdx.x;
    const int m0   = blockIdx.y * BM;
    const int n0   = blockIdx.x * BN;
    const int row  = tid >> 1;
    const int seg  = (tid & 1) * 16;
    const int wave = tid >> 5, lane = tid & 31;
    const int wm = (wave & 3) * 32, wn = (wave >> 2) * 64;
    const int l16 = lane & 15, lhalf = lane >> 4;

    v8f zero8 = {0.f, 0.f, 0.f, 0.f, 0.f, 0.f, 0.f, 0.f};
    v8f acc[2][4];
#pragma unroll
    for (int mi = 0; mi < 2; ++mi)
#pragma unroll
        for (int ni = 0; ni < 4; ++ni) acc[mi][ni] = zero8;

    const int gm = m0 + row;
    const int gn = n0 + row;
    const __bf16* Arow = A + (size_t)gm * K;
    const __bf16* Brow = Bt + (size_t)gn * K;

    auto stage_tile = [&](int p, int k0) {
        int k = k0 + seg;
        stage16(Arow + k,     gm < M, k,     K, &As[p][row * ALD + seg]);
        stage16(Arow + k + 8, gm < M, k + 8, K, &As[p][row * ALD + seg + 8]);
        stage16(Brow + k,     gn < N, k,     K, &Bs[p][row * BLD + seg]);
        stage16(Brow + k + 8, gn < N, k + 8, K, &Bs[p][row * BLD + seg + 8]);
    };

    stage_tile(0, 0);
    ASYNC_WAIT();
    __syncthreads();

    int p = 0;
    for (int k0 = 0; k0 < K; k0 += BKK, p ^= 1) {
        if (k0 + BKK < K) stage_tile(p ^ 1, k0 + BKK);

        Frag af[2], bf[4];
#pragma unroll
        for (int mi = 0; mi < 2; ++mi) {
            const __bf16* ap = &As[p][(wm + mi * 16 + l16) * ALD + lhalf * 8];
            af[mi].f[0] = *(const float4*)ap;
            af[mi].f[1] = *(const float4*)(ap + 16);
        }
#pragma unroll
        for (int ni = 0; ni < 4; ++ni) {
            const __bf16* bp = &Bs[p][(wn + ni * 16 + l16) * BLD + lhalf * 8];
            bf[ni].f[0] = *(const float4*)bp;
            bf[ni].f[1] = *(const float4*)(bp + 16);
        }
#pragma unroll
        for (int mi = 0; mi < 2; ++mi)
#pragma unroll
            for (int ni = 0; ni < 4; ++ni)
                acc[mi][ni] = __builtin_amdgcn_wmma_f32_16x16x32_bf16(
                    false, af[mi].v, false, bf[ni].v, (short)0, acc[mi][ni], false, false);

        ASYNC_WAIT();
        __syncthreads();
    }

#pragma unroll
    for (int mi = 0; mi < 2; ++mi) {
#pragma unroll
        for (int ni = 0; ni < 4; ++ni) {
            int cn = n0 + wn + ni * 16 + l16;
            if (cn >= N) continue;
            int cmb = m0 + wm + mi * 16 + lhalf * 8;
            float bv = (EPI >= 1) ? bias[cn] : 0.f;
#pragma unroll
            for (int r = 0; r < 8; ++r) {
                int cm = cmb + r;
                if (cm >= M) continue;
                float v = acc[mi][ni][r] + bv;
                if (EPI == 2) v = v > 0.f ? v : 0.01f * v;
                C[(size_t)cm * N + cn] = v;
            }
        }
    }
}

// ---------------------------------------------------------------------------
// Fused final lin1: H[E,128] = leaky( [xsrc[src] | eattr | xdst[dst]] @ W1 + b1 )
// A gathered on the fly (K = 2176 = 1024 + 128 + 1024), W1t is [128,2176] bf16.
// Double-buffered LDS + async copies, same WMMA core.
// ---------------------------------------------------------------------------
__global__ __launch_bounds__(256) void edge_lin1_kernel(
    const __bf16* __restrict__ xsrc, const __bf16* __restrict__ xdst,
    const __bf16* __restrict__ eattr, const int* __restrict__ src,
    const int* __restrict__ dst, const __bf16* __restrict__ W1t,
    const float* __restrict__ b1, float* __restrict__ Hout, int E) {
    const int K = 2176;
    __shared__ __bf16 As[2][BM * ALD];
    __shared__ __bf16 Bs[2][BN * BLD];

    const int tid  = threadIdx.x;
    const int m0   = blockIdx.y * BM;
    const int row  = tid >> 1;
    const int seg  = (tid & 1) * 16;
    const int wave = tid >> 5, lane = tid & 31;
    const int wm = (wave & 3) * 32, wn = (wave >> 2) * 64;
    const int l16 = lane & 15, lhalf = lane >> 4;

    v8f zero8 = {0.f, 0.f, 0.f, 0.f, 0.f, 0.f, 0.f, 0.f};
    v8f acc[2][4];
#pragma unroll
    for (int mi = 0; mi < 2; ++mi)
#pragma unroll
        for (int ni = 0; ni < 4; ++ni) acc[mi][ni] = zero8;

    const int e = m0 + row;
    int si = 0, di = 0;
    if (e < E) { si = src[e]; di = dst[e]; }

    auto stage_tile = [&](int p, int k0) {
        int k = k0 + seg;  // 16-aligned chunk: region-uniform (1024,1152 are /16)
        const __bf16* gp;
        if (k < 1024)      gp = xsrc + (size_t)si * 1024 + k;
        else if (k < 1152) gp = eattr + (size_t)e * 128 + (k - 1024);
        else               gp = xdst + (size_t)di * 1024 + (k - 1152);
        stage16_full(gp,     e < E, &As[p][row * ALD + seg]);
        stage16_full(gp + 8, e < E, &As[p][row * ALD + seg + 8]);
        const __bf16* bp = W1t + (size_t)row * K + k;  // row < 128 == N
        stage16_full(bp,     true, &Bs[p][row * BLD + seg]);
        stage16_full(bp + 8, true, &Bs[p][row * BLD + seg + 8]);
    };

    stage_tile(0, 0);
    ASYNC_WAIT();
    __syncthreads();

    int p = 0;
    for (int k0 = 0; k0 < K; k0 += BKK, p ^= 1) {
        if (k0 + BKK < K) stage_tile(p ^ 1, k0 + BKK);

        Frag af[2], bf[4];
#pragma unroll
        for (int mi = 0; mi < 2; ++mi) {
            const __bf16* ap = &As[p][(wm + mi * 16 + l16) * ALD + lhalf * 8];
            af[mi].f[0] = *(const float4*)ap;
            af[mi].f[1] = *(const float4*)(ap + 16);
        }
#pragma unroll
        for (int ni = 0; ni < 4; ++ni) {
            const __bf16* bsp = &Bs[p][(wn + ni * 16 + l16) * BLD + lhalf * 8];
            bf[ni].f[0] = *(const float4*)bsp;
            bf[ni].f[1] = *(const float4*)(bsp + 16);
        }
#pragma unroll
        for (int mi = 0; mi < 2; ++mi)
#pragma unroll
            for (int ni = 0; ni < 4; ++ni)
                acc[mi][ni] = __builtin_amdgcn_wmma_f32_16x16x32_bf16(
                    false, af[mi].v, false, bf[ni].v, (short)0, acc[mi][ni], false, false);

        ASYNC_WAIT();
        __syncthreads();
    }

#pragma unroll
    for (int mi = 0; mi < 2; ++mi) {
#pragma unroll
        for (int ni = 0; ni < 4; ++ni) {
            int cn = wn + ni * 16 + l16;  // < 128 always
            int cmb = m0 + wm + mi * 16 + lhalf * 8;
            float bv = b1[cn];
#pragma unroll
            for (int r = 0; r < 8; ++r) {
                int cm = cmb + r;
                if (cm >= E) continue;
                float v = acc[mi][ni][r] + bv;
                v = v > 0.f ? v : 0.01f * v;
                Hout[(size_t)cm * 128 + cn] = v;
            }
        }
    }
}

// ---------------------------------------------------------------------------
// Small VALU kernels
// ---------------------------------------------------------------------------
__global__ void fill_kernel(float* p, float v, int n) {
    int i = blockIdx.x * blockDim.x + threadIdx.x;
    if (i < n) p[i] = v;
}

__global__ void cvt_bf16_kernel(const float* __restrict__ x, __bf16* __restrict__ y, int n) {
    int i = blockIdx.x * blockDim.x + threadIdx.x;
    if (i < n) y[i] = to_bf16(x[i]);
}

// Wt[n*K+k] = (bf16) W[k*N+n]
__global__ void transpose_bf16_kernel(const float* __restrict__ W, __bf16* __restrict__ Wt,
                                      int K, int N) {
    int i = blockIdx.x * blockDim.x + threadIdx.x;
    if (i < K * N) {
        int k = i / N, n = i - k * N;
        Wt[(size_t)n * K + k] = to_bf16(W[i]);
    }
}

// wa[k*8+h] = sum_c W[k*1024 + h*128 + c] * avec[h*128+c]
__global__ void wa_kernel(const float* __restrict__ W, const float* __restrict__ avec,
                          float* __restrict__ wa, int K) {
    int i = blockIdx.x * blockDim.x + threadIdx.x;
    if (i < K * 8) {
        int k = i >> 3, h = i & 7;
        const float* wr = W + (size_t)k * 1024 + h * 128;
        const float* ar = avec + h * 128;
        float s = 0.f;
        for (int c = 0; c < 128; ++c) s += wr[c] * ar[c];
        wa[i] = s;
    }
}

// aout[n*8+h] = sum_k x[n*K+k] * wa[k*8+h]
__global__ void att_node_kernel(const float* __restrict__ x, const float* __restrict__ wa,
                                float* __restrict__ aout, int Nn, int K) {
    int i = blockIdx.x * blockDim.x + threadIdx.x;
    if (i < Nn * 8) {
        int n = i >> 3, h = i & 7;
        const float* xr = x + (size_t)n * K;
        float s = 0.f;
        for (int k = 0; k < K; ++k) s += xr[k] * wa[k * 8 + h];
        aout[i] = s;
    }
}

__device__ inline void atomicMaxF(float* addr, float val) {
    unsigned int* ia = (unsigned int*)addr;
    unsigned int cur = *ia;
    while (__uint_as_float(cur) < val) {
        unsigned int prev = atomicCAS(ia, cur, __float_as_uint(val));
        if (prev == cur) break;
        cur = prev;
    }
}

__global__ void edge_logit_max_kernel(const int* __restrict__ src, const int* __restrict__ dst,
                                      const float* __restrict__ as_, const float* __restrict__ ad_,
                                      float* __restrict__ eb, float* __restrict__ m, int E) {
    int i = blockIdx.x * blockDim.x + threadIdx.x;
    if (i < E * 8) {
        int e = i >> 3, h = i & 7;
        float v = as_[src[e] * 8 + h] + ad_[dst[e] * 8 + h];
        v = v > 0.f ? v : 0.2f * v;  // ATT_SLOPE
        eb[i] = v;
        atomicMaxF(&m[dst[e] * 8 + h], v);
    }
}

__global__ void edge_exp_kernel(const int* __restrict__ dst, float* __restrict__ eb,
                                const float* __restrict__ m, float* __restrict__ denom, int E) {
    int i = blockIdx.x * blockDim.x + threadIdx.x;
    if (i < E * 8) {
        int e = i >> 3, h = i & 7;
        float ex = __expf(eb[i] - m[dst[e] * 8 + h]);
        eb[i] = ex;
        atomicAdd(&denom[dst[e] * 8 + h], ex);
    }
}

// one block per edge; 256 threads x 4 channels = 1024 channels
__global__ void scatter_kernel(const float* __restrict__ hs, const int* __restrict__ src,
                               const int* __restrict__ dst, const float* __restrict__ ex,
                               const float* __restrict__ denom, float* __restrict__ acc,
                               float scale, int E) {
    int e = blockIdx.x;
    if (e >= E) return;
    int c0 = threadIdx.x * 4;
    int s = src[e], d = dst[e];
    int h = c0 >> 7;
    float w = ex[(size_t)e * 8 + h] / (denom[d * 8 + h] + 1e-16f) * scale;
    const float* hp = hs + (size_t)s * 1024 + c0;
    float* ap = acc + (size_t)d * 1024 + c0;
    atomicAdd(ap + 0, w * hp[0]);
    atomicAdd(ap + 1, w * hp[1]);
    atomicAdd(ap + 2, w * hp[2]);
    atomicAdd(ap + 3, w * hp[3]);
}

__global__ void combine_kernel(const float* __restrict__ acc, const float* __restrict__ ba,
                               const float* __restrict__ bb, float scale,
                               const float* __restrict__ xold, int residual,
                               float* __restrict__ x, __bf16* __restrict__ xbf, int n) {
    int i = blockIdx.x * blockDim.x + threadIdx.x;
    if (i < n) {
        int c = i & 1023;
        float v = acc[i] + scale * (ba[c] + (bb ? bb[c] : 0.f));
        if (residual) v += xold[i];
        v = v > 0.f ? v : 0.01f * v;  // SLOPE
        x[i] = v;
        xbf[i] = to_bf16(v);
    }
}

// eattr_bf[e*128+j] = bf16( sum_k ea[e*16+k]*W[k*128+j] + b[j] )
__global__ void eattr_kernel(const float* __restrict__ ea, const float* __restrict__ W,
                             const float* __restrict__ b, __bf16* __restrict__ out, int E) {
    int i = blockIdx.x * blockDim.x + threadIdx.x;
    if (i < E * 128) {
        int e = i >> 7, j = i & 127;
        const float* er = ea + (size_t)e * 16;
        float s = b[j];
        for (int k = 0; k < 16; ++k) s += er[k] * W[k * 128 + j];
        out[i] = to_bf16(s);
    }
}

__global__ void lin2_kernel(const float* __restrict__ Hm, const float* __restrict__ w2,
                            const float* __restrict__ b2, float* __restrict__ out, int E) {
    int e = blockIdx.x * blockDim.x + threadIdx.x;
    if (e < E) {
        const float* hr = Hm + (size_t)e * 128;
        float s = b2[0];
        for (int j = 0; j < 128; ++j) s += hr[j] * w2[j];
        out[e] = s;
    }
}

// ---------------------------------------------------------------------------
// Host orchestration
// ---------------------------------------------------------------------------
struct GatBufs {
    __bf16* wt; float* hs; float* wa; float* a_s; float* a_d;
    float* mbuf; float* dbuf; float* eb;
};

static void run_gat(hipStream_t st, const GatBufs& B,
                    const float* Ws, const float* Wd, const float* avS, const float* avD,
                    const float* xsrc_f, const __bf16* xsrc_b, int Ns, int Ksrc,
                    const float* xdst_f, int Nd, int Kdst,
                    const int* src, const int* dst, int E, float scale, float* acc) {
    transpose_bf16_kernel<<<cdiv(Ksrc * 1024, 256), 256, 0, st>>>(Ws, B.wt, Ksrc, 1024);
    gemm_bf16_kernel<0><<<dim3(1024 / BN, cdiv(Ns, BM)), 256, 0, st>>>(
        xsrc_b, B.wt, B.hs, Ns, 1024, Ksrc, nullptr);
    wa_kernel<<<cdiv(Ksrc * 8, 256), 256, 0, st>>>(Ws, avS, B.wa, Ksrc);
    att_node_kernel<<<cdiv(Ns * 8, 256), 256, 0, st>>>(xsrc_f, B.wa, B.a_s, Ns, Ksrc);
    wa_kernel<<<cdiv(Kdst * 8, 256), 256, 0, st>>>(Wd, avD, B.wa, Kdst);
    att_node_kernel<<<cdiv(Nd * 8, 256), 256, 0, st>>>(xdst_f, B.wa, B.a_d, Nd, Kdst);
    fill_kernel<<<cdiv(Nd * 8, 256), 256, 0, st>>>(B.mbuf, -3.0e38f, Nd * 8);
    fill_kernel<<<cdiv(Nd * 8, 256), 256, 0, st>>>(B.dbuf, 0.f, Nd * 8);
    edge_logit_max_kernel<<<cdiv(E * 8, 256), 256, 0, st>>>(src, dst, B.a_s, B.a_d, B.eb, B.mbuf, E);
    edge_exp_kernel<<<cdiv(E * 8, 256), 256, 0, st>>>(dst, B.eb, B.mbuf, B.dbuf, E);
    scatter_kernel<<<E, 256, 0, st>>>(B.hs, src, dst, B.eb, B.dbuf, acc, scale, E);
}

extern "C" void kernel_launch(void* const* d_in, const int* in_sizes, int n_in,
                              void* d_out, int out_size, void* d_ws, size_t ws_size,
                              hipStream_t stream) {
    (void)in_sizes; (void)n_in; (void)out_size; (void)ws_size;
    const int NR = 8000, NV = 800;
    const int ERR = 64000, EVR = 24000, ERV = 24000;

    const float* x_req = (const float*)d_in[0];
    const float* x_veh = (const float*)d_in[1];
    const int* rr_s = (const int*)d_in[2];  const int* rr_d = (const int*)d_in[3];
    const int* vr_s = (const int*)d_in[4];  const int* vr_d = (const int*)d_in[5];
    const int* rv_s = (const int*)d_in[6];  const int* rv_d = (const int*)d_in[7];
    const float* ea_rr = (const float*)d_in[8];
    const float* ea_vr = (const float*)d_in[9];
    const float* ea_rv = (const float*)d_in[10];
    // params flattened (sorted dict keys): layers -> lin1 -> lin2 -> proj
    auto P = [&](int i) { return (const float*)d_in[i]; };
    // layers[0]: rr{Wd11,Ws12,ad13,as14,b15} rv{16..20} vr{21..25}
    // layers[1]: rr{26..30} rv{31..35} vr{36..40}
    // lin1 W=41 b=42 ; lin2 W=43 b=44 ; proj rr{45,46} rv{47,48} vr{49,50}
    float* out = (float*)d_out;

    char* ws = (char*)d_ws;
    size_t off = 0;
    auto alloc = [&](size_t bytes) -> void* {
        void* p = ws + off;
        off = (off + bytes + 255) & ~(size_t)255;
        return p;
    };
    float*  xr      = (float*)alloc((size_t)NR * 1024 * 4);
    float*  xv      = (float*)alloc((size_t)NV * 1024 * 4);
    __bf16* xr_bf   = (__bf16*)alloc((size_t)NR * 1024 * 2);
    __bf16* xv_bf   = (__bf16*)alloc((size_t)NV * 1024 * 2);
    __bf16* xinr_bf = (__bf16*)alloc((size_t)NR * 32 * 2);
    __bf16* xinv_bf = (__bf16*)alloc((size_t)NV * 16 * 2);
    float*  acc_r   = (float*)alloc((size_t)NR * 1024 * 4);
    float*  acc_v   = (float*)alloc((size_t)NV * 1024 * 4);
    GatBufs B;
    B.wt   = (__bf16*)alloc((size_t)1024 * 2176 * 2);
    B.hs   = (float*)alloc((size_t)NR * 1024 * 4);
    B.wa   = (float*)alloc((size_t)1024 * 8 * 4);
    B.a_s  = (float*)alloc((size_t)NR * 8 * 4);
    B.a_d  = (float*)alloc((size_t)NR * 8 * 4);
    B.mbuf = (float*)alloc((size_t)NR * 8 * 4);
    B.dbuf = (float*)alloc((size_t)NR * 8 * 4);
    B.eb   = (float*)alloc((size_t)ERR * 8 * 4);
    __bf16* eattr_bf = (__bf16*)alloc((size_t)ERR * 128 * 2);
    float*  hlin     = (float*)alloc((size_t)ERR * 128 * 4);

    hipStream_t st = stream;

    // ---- layer 1 ----
    cvt_bf16_kernel<<<cdiv(NR * 32, 256), 256, 0, st>>>(x_req, xinr_bf, NR * 32);
    cvt_bf16_kernel<<<cdiv(NV * 16, 256), 256, 0, st>>>(x_veh, xinv_bf, NV * 16);
    fill_kernel<<<cdiv(NR * 1024, 256), 256, 0, st>>>(acc_r, 0.f, NR * 1024);
    fill_kernel<<<cdiv(NV * 1024, 256), 256, 0, st>>>(acc_v, 0.f, NV * 1024);
    run_gat(st, B, P(12), P(11), P(14), P(13), x_req, xinr_bf, NR, 32,
            x_req, NR, 32, rr_s, rr_d, ERR, 0.5f, acc_r);                 // rr
    run_gat(st, B, P(22), P(21), P(24), P(23), x_veh, xinv_bf, NV, 16,
            x_req, NR, 32, vr_s, vr_d, EVR, 0.5f, acc_r);                 // vr
    run_gat(st, B, P(17), P(16), P(19), P(18), x_req, xinr_bf, NR, 32,
            x_veh, NV, 16, rv_s, rv_d, ERV, 1.0f, acc_v);                 // rv
    combine_kernel<<<cdiv(NR * 1024, 256), 256, 0, st>>>(acc_r, P(15), P(25), 0.5f,
                                                         nullptr, 0, xr, xr_bf, NR * 1024);
    combine_kernel<<<cdiv(NV * 1024, 256), 256, 0, st>>>(acc_v, P(20), nullptr, 1.0f,
                                                         nullptr, 0, xv, xv_bf, NV * 1024);

    // ---- layer 2 ----
    fill_kernel<<<cdiv(NR * 1024, 256), 256, 0, st>>>(acc_r, 0.f, NR * 1024);
    fill_kernel<<<cdiv(NV * 1024, 256), 256, 0, st>>>(acc_v, 0.f, NV * 1024);
    run_gat(st, B, P(27), P(26), P(29), P(28), xr, xr_bf, NR, 1024,
            xr, NR, 1024, rr_s, rr_d, ERR, 0.5f, acc_r);                  // rr
    run_gat(st, B, P(37), P(36), P(39), P(38), xv, xv_bf, NV, 1024,
            xr, NR, 1024, vr_s, vr_d, EVR, 0.5f, acc_r);                  // vr
    run_gat(st, B, P(32), P(31), P(34), P(33), xr, xr_bf, NR, 1024,
            xv, NV, 1024, rv_s, rv_d, ERV, 1.0f, acc_v);                  // rv
    combine_kernel<<<cdiv(NR * 1024, 256), 256, 0, st>>>(acc_r, P(30), P(40), 0.5f,
                                                         xr, 1, xr, xr_bf, NR * 1024);
    combine_kernel<<<cdiv(NV * 1024, 256), 256, 0, st>>>(acc_v, P(35), nullptr, 1.0f,
                                                         xv, 1, xv, xv_bf, NV * 1024);

    // ---- final per-edge MLP ----
    __bf16* w1t = B.wt;  // reuse: [128, 2176] bf16
    transpose_bf16_kernel<<<cdiv(2176 * 128, 256), 256, 0, st>>>(P(41), w1t, 2176, 128);
    // rr
    eattr_kernel<<<cdiv(ERR * 128, 256), 256, 0, st>>>(ea_rr, P(45), P(46), eattr_bf, ERR);
    edge_lin1_kernel<<<dim3(1, cdiv(ERR, BM)), 256, 0, st>>>(xr_bf, xr_bf, eattr_bf,
                                                             rr_s, rr_d, w1t, P(42), hlin, ERR);
    lin2_kernel<<<cdiv(ERR, 256), 256, 0, st>>>(hlin, P(43), P(44), out, ERR);
    // vr
    eattr_kernel<<<cdiv(EVR * 128, 256), 256, 0, st>>>(ea_vr, P(49), P(50), eattr_bf, EVR);
    edge_lin1_kernel<<<dim3(1, cdiv(EVR, BM)), 256, 0, st>>>(xv_bf, xr_bf, eattr_bf,
                                                             vr_s, vr_d, w1t, P(42), hlin, EVR);
    lin2_kernel<<<cdiv(EVR, 256), 256, 0, st>>>(hlin, P(43), P(44), out + ERR, EVR);
    // rv
    eattr_kernel<<<cdiv(ERV * 128, 256), 256, 0, st>>>(ea_rv, P(47), P(48), eattr_bf, ERV);
    edge_lin1_kernel<<<dim3(1, cdiv(ERV, BM)), 256, 0, st>>>(xr_bf, xv_bf, eattr_bf,
                                                             rv_s, rv_d, w1t, P(42), hlin, ERV);
    lin2_kernel<<<cdiv(ERV, 256), 256, 0, st>>>(hlin, P(43), P(44), out + ERR + EVR, ERV);
}
